// VectorQuantizer_46686294508024
// MI455X (gfx1250) — matrix operand order repaired and checked by reading, not compiled, not run on hardware
//
#include <hip/hip_runtime.h>

// ---------- types ----------
typedef __attribute__((ext_vector_type(16))) __bf16 v16bf;
typedef __attribute__((ext_vector_type(8)))  float  v8f;
typedef __attribute__((ext_vector_type(4)))  unsigned int u32x4;

union FragU {
    unsigned int u[8];
    u32x4 q[2];
    v16bf v;
};

// fp32 -> bf16 round-to-nearest-even (bits)
__device__ __forceinline__ unsigned short f2bf(float f) {
    unsigned int u = __float_as_uint(f);
    unsigned int r = u + 0x7FFFu + ((u >> 16) & 1u);
    return (unsigned short)(r >> 16);
}

#define N_EMBED 1024
#define EMB_DIM 256
#define N_ROWS  32768               // 32 * 32 * 32
#define ROWS_PER_WG 64
#define ROW_PAD 264                 // bf16 per padded LDS row (528B, 16B aligned, stride%64dw==4)
#define CHUNK_CODES 128

// ---------------- prepass: bf16 codebook + ||e||^2 + zero acc ----------------
__global__ __launch_bounds__(256) void vq_prep_kernel(
    const float* __restrict__ emb, unsigned short* __restrict__ embbf,
    float* __restrict__ enorm, float* __restrict__ acc)
{
    const int code = blockIdx.x;
    const int t = threadIdx.x;
    float v = emb[code * EMB_DIM + t];
    embbf[code * EMB_DIM + t] = f2bf(v);
    float s = v * v;
    __shared__ float wsum[8];
    #pragma unroll
    for (int off = 16; off >= 1; off >>= 1) s += __shfl_xor(s, off, 32);
    if ((t & 31) == 0) wsum[t >> 5] = s;
    __syncthreads();
    if (t < 8) {
        float x = wsum[t];
        #pragma unroll
        for (int off = 4; off >= 1; off >>= 1) x += __shfl_xor(x, off, 32);
        if (t == 0) enorm[code] = x;
    }
    if (code == 0 && t == 0) *acc = 0.0f;
}

// ---------------- main: WMMA distance argmin + gather + loss ----------------
__global__ __launch_bounds__(128) void vq_main_kernel(
    const float* __restrict__ z, const float* __restrict__ emb,
    const unsigned short* __restrict__ embbf, const float* __restrict__ enorm,
    float* __restrict__ out, float* __restrict__ acc)
{
    __shared__ unsigned short lds_z[ROWS_PER_WG * ROW_PAD];   // 33792 B
    __shared__ unsigned short lds_e[CHUNK_CODES * ROW_PAD];   // 67584 B
    __shared__ int   idx_buf[ROWS_PER_WG];
    __shared__ float red[4];

    const int tid  = threadIdx.x;
    const int wave = tid >> 5;
    const int lane = tid & 31;
    const int m    = lane & 15;
    const int hi   = lane >> 4;                  // half-wave id

    const int row_base = blockIdx.x * ROWS_PER_WG;   // global flat row n
    const int b  = row_base >> 10;                   // batch
    const int hw = row_base & 1023;                  // h*32+w base (rows contiguous here)
    const size_t zbase = (size_t)b * (EMB_DIM * 1024) + (size_t)hw;  // + d*1024 + r

    // ---- stage z tile [64 rows x 256 d] as bf16 into LDS (coalesced global reads)
    #pragma unroll 4
    for (int i = 0; i < 128; ++i) {
        int f = i * 128 + tid;       // 0..16383
        int d = f >> 6;
        int r = f & 63;
        float v = z[zbase + (size_t)d * 1024 + (size_t)r];
        lds_z[r * ROW_PAD + d] = f2bf(v);
    }
    __syncthreads();

    // ---- build 8 A fragments (reused for every code tile)
    // 16-bit A layout: lane<16 -> K {kb+0..7, kb+16..23}; lane>=16 -> K {kb+8..15, kb+24..31}
    const unsigned short* zrow = &lds_z[((wave << 4) + m) * ROW_PAD];
    v16bf afrag[8];
    #pragma unroll
    for (int ks = 0; ks < 8; ++ks) {
        const int kb = ks * 32;
        FragU a;
        a.q[0] = *(const u32x4*)(zrow + kb + (hi ? 8 : 0));
        a.q[1] = *(const u32x4*)(zrow + kb + (hi ? 24 : 16));
        afrag[ks] = a.v;
    }

    float best[8];
    int   bidx[8];
    #pragma unroll
    for (int r = 0; r < 8; ++r) { best[r] = 3.402823e38f; bidx[r] = 0; }

    // ---- sweep all 1024 codes in chunks of 128 staged in LDS
    for (int chunk = 0; chunk < N_EMBED / CHUNK_CODES; ++chunk) {
        __syncthreads();   // previous chunk fully consumed
        const unsigned int* src =
            (const unsigned int*)embbf + (size_t)chunk * CHUNK_CODES * (EMB_DIM / 2);
        unsigned int* dst = (unsigned int*)lds_e;
        #pragma unroll 4
        for (int i = 0; i < 128; ++i) {
            int f = i * 128 + tid;       // 0..16383 dwords
            int code = f >> 7;           // 128 dwords per code row
            int off  = f & 127;
            dst[code * (ROW_PAD / 2) + off] = src[f];
        }
        __syncthreads();

        #pragma unroll 1
        for (int ct = 0; ct < CHUNK_CODES / 16; ++ct) {
            // B layout (32x16): lane<16 -> n=lane, K {kb+0..7, kb+16..23} ; lane>=16 -> n=lane-16, K {kb+8..15, kb+24..31}
            const unsigned short* erow = &lds_e[(ct * 16 + m) * ROW_PAD];
            v8f c = {};
            #pragma unroll
            for (int ks = 0; ks < 8; ++ks) {
                FragU bf;
                const unsigned short* p = erow + ks * 32;
                bf.q[0] = *(const u32x4*)(p + (hi ? 8 : 0));
                bf.q[1] = *(const u32x4*)(p + (hi ? 24 : 16));
                c = __builtin_amdgcn_wmma_f32_16x16x32_bf16(
                        false, afrag[ks], false, bf.v, (short)0, c, false, false);
            }
            const int code = chunk * CHUNK_CODES + ct * 16 + m;
            const float en = enorm[code];
            #pragma unroll
            for (int r = 0; r < 8; ++r) {
                float s = en - 2.0f * c[r];
                if (s < best[r]) { best[r] = s; bidx[r] = code; }
            }
        }
    }

    // ---- argmin across the 16 lanes of each half-wave (ties -> smaller index)
    #pragma unroll
    for (int r = 0; r < 8; ++r) {
        float v = best[r];
        int   ix = bidx[r];
        #pragma unroll
        for (int off = 8; off >= 1; off >>= 1) {
            float ov = __shfl_xor(v, off, 32);
            int   oi = __shfl_xor(ix, off, 32);
            if (ov < v || (ov == v && oi < ix)) { v = ov; ix = oi; }
        }
        if (m == 0) idx_buf[(wave << 4) + (hi << 3) + r] = ix;
    }
    __syncthreads();

    // ---- writeback z_st (== z_quant) in NCHW + loss accumulation
    float sq = 0.0f;
    #pragma unroll 4
    for (int i = 0; i < 128; ++i) {
        int f = i * 128 + tid;
        int d = f >> 6;
        int r = f & 63;
        size_t go = zbase + (size_t)d * 1024 + (size_t)r;
        float q  = emb[idx_buf[r] * EMB_DIM + d];
        float zv = z[go];
        out[go] = q;
        float df = q - zv;
        sq += df * df;
    }
    #pragma unroll
    for (int off = 16; off >= 1; off >>= 1) sq += __shfl_xor(sq, off, 32);
    if (lane == 0) red[wave] = sq;
    __syncthreads();
    if (tid == 0) {
        float t = red[0] + red[1] + red[2] + red[3];
        atomicAdd(acc, t);
    }
}

// ---------------- finalize scalars ----------------
__global__ void vq_fin_kernel(const float* __restrict__ acc, float* __restrict__ out3)
{
    float mean = *acc * (1.0f / ((float)N_ROWS * (float)EMB_DIM));
    out3[0] = 1.25f * mean;  // loss = codebook + 0.25*commitment (equal values)
    out3[1] = mean;          // codebook_loss
    out3[2] = mean;          // commitment_loss
}

extern "C" void kernel_launch(void* const* d_in, const int* in_sizes, int n_in,
                              void* d_out, int out_size, void* d_ws, size_t ws_size,
                              hipStream_t stream) {
    const float* z   = (const float*)d_in[0];   // [32,256,32,32] fp32
    const float* emb = (const float*)d_in[1];   // [1024,256] fp32
    float* out = (float*)d_out;                 // 8388608 z_st + 3 scalars

    char* ws = (char*)d_ws;
    float*          acc   = (float*)ws;                       // 4 B
    float*          enorm = (float*)(ws + 256);               // 4 KB
    unsigned short* embbf = (unsigned short*)(ws + 8192);     // 512 KB bf16 codebook

    vq_prep_kernel<<<N_EMBED, EMB_DIM, 0, stream>>>(emb, embbf, enorm, acc);
    vq_main_kernel<<<N_ROWS / ROWS_PER_WG, 128, 0, stream>>>(z, emb, embbf, enorm, out, acc);
    vq_fin_kernel<<<1, 1, 0, stream>>>(acc, out + 8388608);
}
